// LatentInference_5875515261562
// MI455X (gfx1250) — compile-verified, matrix operand order checked by reading, not run on hardware
//
#include <hip/hip_runtime.h>
#include <hip/hip_bf16.h>

// ---------------------------------------------------------------------------
// Types for CDNA5 WMMA (gfx1250, wave32): D(f32 16x16) = A(bf16 16x32) x B(bf16 32x16) + C
// ---------------------------------------------------------------------------
typedef __attribute__((ext_vector_type(16))) __bf16          bf16x16;
typedef __attribute__((ext_vector_type(8)))  float           f32x8;
typedef __attribute__((ext_vector_type(8)))  unsigned short  u16x8;
typedef __attribute__((ext_vector_type(16))) unsigned short  u16x16;

static __device__ __forceinline__ unsigned short f2bf(float f) {
    unsigned int u = __float_as_uint(f);
    u += 0x7FFFu + ((u >> 16) & 1u);           // round-to-nearest-even
    return (unsigned short)(u >> 16);
}

// Load a 16-element bf16 fragment for A or B operand.
// `base` points at elements [K..K+7] for this lane; [K+16..K+23] at base+16
// (ISA 7.12.2 16-bit A/B layout; caller folds half*8 into base).
static __device__ __forceinline__ bf16x16 load_frag(const unsigned short* base) {
    u16x8 lo = *reinterpret_cast<const u16x8*>(base);
    u16x8 hi = *reinterpret_cast<const u16x8*>(base + 16);
    u16x16 full;
#pragma unroll
    for (int i = 0; i < 8; ++i) { full[i] = lo[i]; full[i + 8] = hi[i]; }
    return __builtin_bit_cast(bf16x16, full);
}

static __device__ __forceinline__ f32x8 wmma_bf16(bf16x16 a, bf16x16 b, f32x8 c) {
    return __builtin_amdgcn_wmma_f32_16x16x32_bf16(
        /*neg_a=*/false, a, /*neg_b=*/false, b,
        /*c_mod=*/(short)0, c, /*reuse_a=*/false, /*reuse_b=*/false);
}

// ---------------------------------------------------------------------------
// Generic GEMM: C[M,NT] = act(alpha * (A[M,K] @ W[K,NT]))
// fp32 in / fp32 out; bf16 WMMA internally. K % 32 == 0. NT compile-time.
// Block = 128 threads = 4 waves; block computes 64 rows x NT cols.
// ---------------------------------------------------------------------------
template <int NT>
__global__ void __launch_bounds__(128)
gemm_wmma(const float* __restrict__ A, const float* __restrict__ W,
          float* __restrict__ C, int M, int K, int act, float alpha)
{
    constexpr int NTILES = NT / 16;
    __shared__ __align__(16) unsigned short lA[64 * 32];   // A tile, row-major [row][k]
    __shared__ __align__(16) unsigned short lB[NT * 32];   // W tile transposed [n][k]

    const int tid  = threadIdx.x;
    const int lane = tid & 31;
    const int wave = tid >> 5;
    const int rowbase = blockIdx.x * 64;
    const int m    = lane & 15;
    const int half = lane >> 4;

    f32x8 acc[NTILES];
#pragma unroll
    for (int t = 0; t < NTILES; ++t) acc[t] = f32x8{0, 0, 0, 0, 0, 0, 0, 0};

    for (int kk = 0; kk < K; kk += 32) {
        // --- stage A tile (64x32 fp32 -> bf16), 512 float4 loads over 128 thr
#pragma unroll
        for (int i = 0; i < 4; ++i) {
            int idx4 = tid + i * 128;            // 0..511
            int r    = idx4 >> 3;                // 8 float4 per row
            int c4   = idx4 & 7;
            int grow = rowbase + r;
            float4 v = make_float4(0.f, 0.f, 0.f, 0.f);
            if (grow < M) v = *(const float4*)(A + (size_t)grow * K + kk + c4 * 4);
            ushort4 b;
            b.x = f2bf(v.x); b.y = f2bf(v.y); b.z = f2bf(v.z); b.w = f2bf(v.w);
            *(ushort4*)(&lA[r * 32 + c4 * 4]) = b;
        }
        // --- stage W tile transposed (32 x NT fp32 -> bf16 [n][k])
#pragma unroll
        for (int i = 0; i < (32 * NT) / 128; ++i) {
            int idx = tid + i * 128;
            int k = idx / NT;
            int n = idx & (NT - 1);              // consecutive tid -> consecutive n
            lB[n * 32 + k] = f2bf(W[(size_t)(kk + k) * NT + n]);
        }
        // --- prefetch next A tile rows (global_prefetch_b8)
        if (kk + 32 < K) {
            int grow = rowbase + (tid & 63);
            if (grow < M) __builtin_prefetch(A + (size_t)grow * K + kk + 32, 0, 1);
        }
        __syncthreads();

        bf16x16 af = load_frag(&lA[(wave * 16 + m) * 32 + half * 8]);
#pragma unroll
        for (int nt = 0; nt < NTILES; ++nt) {
            bf16x16 bfr = load_frag(&lB[(nt * 16 + m) * 32 + half * 8]);
            acc[nt] = wmma_bf16(af, bfr, acc[nt]);
        }
        __syncthreads();
    }

    // --- store, C layout: lane 0-15 -> N=lane, M=v ; lane 16-31 -> N=lane-16, M=v+8
#pragma unroll
    for (int nt = 0; nt < NTILES; ++nt) {
#pragma unroll
        for (int v = 0; v < 8; ++v) {
            int grow = rowbase + wave * 16 + (half ? v + 8 : v);
            if (grow < M) {
                float x = acc[nt][v] * alpha;
                if (act) x = (x > 0.f) ? x : 0.01f * x;   // leaky_relu(0.01)
                C[(size_t)grow * NT + nt * 16 + m] = x;
            }
        }
    }
}

// ---------------------------------------------------------------------------
// COO SpMM: out[rows[e], :] += vals[e] * x[cols[e], :]   (d % 4 == 0)
// One wave per edge per iteration; float4 gathers + f32 atomics.
// ---------------------------------------------------------------------------
__global__ void __launch_bounds__(256)
spmm_kernel(const int* __restrict__ rows, const int* __restrict__ cols,
            const float* __restrict__ vals, const float* __restrict__ x,
            float* __restrict__ out, int E, int d)
{
    const int gid  = blockIdx.x * blockDim.x + threadIdx.x;
    const int wid  = gid >> 5;
    const int lane = gid & 31;
    const int nw   = (gridDim.x * blockDim.x) >> 5;
    const int d4   = d >> 2;

    for (int e = wid; e < E; e += nw) {
        const int r = rows[e];
        const int c = cols[e];
        const float v = vals[e];
        const float4* xr = (const float4*)(x + (size_t)c * d);
        float* orow = out + (size_t)r * d;
        for (int j = lane; j < d4; j += 32) {
            float4 g = xr[j];
            atomicAdd(orow + j * 4 + 0, g.x * v);
            atomicAdd(orow + j * 4 + 1, g.y * v);
            atomicAdd(orow + j * 4 + 2, g.z * v);
            atomicAdd(orow + j * 4 + 3, g.w * v);
        }
    }
}

__global__ void __launch_bounds__(256)
leaky_kernel(float* __restrict__ p, size_t n)
{
    size_t i  = (size_t)blockIdx.x * blockDim.x + threadIdx.x;
    size_t st = (size_t)gridDim.x * blockDim.x;
    for (; i < n; i += st) {
        float x = p[i];
        p[i] = (x > 0.f) ? x : 0.01f * x;
    }
}

// ---------------------------------------------------------------------------
// Attention: ctx[N,64] from Q[N,64] (pre-scaled), K[128,64], V[128,64], 2 heads x dh=32.
// Block = 64 threads = 2 waves; block handles 32 query rows (16 per wave).
// Logits and context via bf16 WMMA; softmax over 128 keys in LDS.
// ---------------------------------------------------------------------------
__global__ void __launch_bounds__(64)
attn_kernel(const float* __restrict__ Q, const float* __restrict__ Kc,
            const float* __restrict__ Vc, float* __restrict__ ctx, int M)
{
    __shared__ __align__(16) unsigned short lK[128 * 64];     // [key][dim]  bf16
    __shared__ __align__(16) unsigned short lV[64 * 128];     // [dim][key]  bf16 (transposed)
    __shared__ __align__(16) unsigned short lQ[2][16 * 64];   // per-wave Q tile bf16
    __shared__ __align__(16) float          lL[2][16 * 128];  // per-wave logits f32
    __shared__ __align__(16) unsigned short lW[2][16 * 128];  // per-wave softmax weights bf16

    const int tid  = threadIdx.x;
    const int lane = tid & 31;
    const int wave = tid >> 5;
    const int rowbase = blockIdx.x * 32 + wave * 16;
    const int m    = lane & 15;
    const int half = lane >> 4;

    // stage K: 128x64 fp32 -> bf16 row-major (2048 float4 over 64 threads)
#pragma unroll 4
    for (int i = 0; i < 32; ++i) {
        int idx4 = tid + i * 64;
        int key  = idx4 >> 4;
        int c4   = idx4 & 15;
        float4 v = *(const float4*)(Kc + key * 64 + c4 * 4);
        ushort4 b;
        b.x = f2bf(v.x); b.y = f2bf(v.y); b.z = f2bf(v.z); b.w = f2bf(v.w);
        *(ushort4*)(&lK[key * 64 + c4 * 4]) = b;
    }
    // stage V transposed: Vt[dim][key]
#pragma unroll 4
    for (int i = 0; i < 128; ++i) {
        int idx = tid + i * 64;
        int key = idx >> 6;
        int dim = idx & 63;
        lV[dim * 128 + key] = f2bf(Vc[idx]);
    }
    // stage this wave's Q tile (16 rows x 64 dims)
#pragma unroll
    for (int i = 0; i < 8; ++i) {
        int idx4 = lane + i * 32;
        int r    = idx4 >> 4;
        int c4   = idx4 & 15;
        int grow = rowbase + r;
        float4 v = make_float4(0.f, 0.f, 0.f, 0.f);
        if (grow < M) v = *(const float4*)(Q + (size_t)grow * 64 + c4 * 4);
        ushort4 b;
        b.x = f2bf(v.x); b.y = f2bf(v.y); b.z = f2bf(v.z); b.w = f2bf(v.w);
        *(ushort4*)(&lQ[wave][r * 64 + c4 * 4]) = b;
    }
    __syncthreads();

#pragma unroll
    for (int h = 0; h < 2; ++h) {
        // logits[16,128] = Qh[16,32] @ Kh^T[32,128]
        bf16x16 af = load_frag(&lQ[wave][m * 64 + h * 32 + half * 8]);
#pragma unroll
        for (int nt = 0; nt < 8; ++nt) {
            bf16x16 bfr = load_frag(&lK[(nt * 16 + m) * 64 + h * 32 + half * 8]);
            f32x8 z = f32x8{0, 0, 0, 0, 0, 0, 0, 0};
            f32x8 lg = wmma_bf16(af, bfr, z);
#pragma unroll
            for (int v = 0; v < 8; ++v)
                lL[wave][(half ? v + 8 : v) * 128 + nt * 16 + m] = lg[v];
        }
        __syncthreads();

        // softmax over 128 keys; 2 lanes per row
        {
            int r    = lane >> 1;
            int hcol = (lane & 1) * 64;
            float* Lrow = &lL[wave][r * 128 + hcol];
            float mx = -1e30f;
#pragma unroll 8
            for (int j = 0; j < 64; ++j) mx = fmaxf(mx, Lrow[j]);
            mx = fmaxf(mx, __shfl_xor(mx, 1, 32));
            float sum = 0.f;
#pragma unroll 8
            for (int j = 0; j < 64; ++j) {
                float e = __expf(Lrow[j] - mx);
                Lrow[j] = e;
                sum += e;
            }
            sum += __shfl_xor(sum, 1, 32);
            float inv = 1.0f / sum;
            unsigned short* Wrow = &lW[wave][r * 128 + hcol];
#pragma unroll 8
            for (int j = 0; j < 64; ++j) Wrow[j] = f2bf(Lrow[j] * inv);
        }
        __syncthreads();

        // ctx[16,32] = w[16,128] @ Vh[128,32]   (K=128 in 4 chunks of 32)
#pragma unroll
        for (int nt = 0; nt < 2; ++nt) {
            f32x8 cacc = f32x8{0, 0, 0, 0, 0, 0, 0, 0};
#pragma unroll
            for (int kc = 0; kc < 4; ++kc) {
                bf16x16 wf = load_frag(&lW[wave][m * 128 + kc * 32 + half * 8]);
                bf16x16 vf = load_frag(&lV[(h * 32 + nt * 16 + m) * 128 + kc * 32 + half * 8]);
                cacc = wmma_bf16(wf, vf, cacc);
            }
#pragma unroll
            for (int v = 0; v < 8; ++v) {
                int grow = rowbase + (half ? v + 8 : v);
                if (grow < M)
                    ctx[(size_t)grow * 64 + h * 32 + nt * 16 + m] = cacc[v];
            }
        }
        __syncthreads();   // lL / lW reused next head
    }
}

// ---------------------------------------------------------------------------
// Host-side orchestration
// ---------------------------------------------------------------------------
extern "C" void kernel_launch(void* const* d_in, const int* in_sizes, int n_in,
                              void* d_out, int out_size, void* d_ws, size_t ws_size,
                              hipStream_t stream)
{
    const float* ns_emb   = (const float*)d_in[0];
    const int*   erows    = (const int*)d_in[1];
    const int*   ecols    = (const int*)d_in[2];
    const float* evals    = (const float*)d_in[3];
    const float* cond     = (const float*)d_in[4];   // [128,128]
    const float* W_hidden = (const float*)d_in[5];   // [256,128]
    const float* W_mu     = (const float*)d_in[6];   // [128,64]
    const float* W_var    = (const float*)d_in[7];   // [128,64]
    const float* Wq       = (const float*)d_in[8];   // [128,64]
    const float* Wk       = (const float*)d_in[9];   // [128,64]
    const float* Wv       = (const float*)d_in[10];  // [128,64]
    const float* Wo       = (const float*)d_in[11];  // [64,128]

    const int N = in_sizes[0] / 256;   // 50000
    const int E = in_sizes[1];         // 600000
    const int L = 128;                 // condition length

    // workspace layout (floats)
    float* bufA = (float*)d_ws;                    // [N,128] support1 -> later support_mu | support_var
    float* bufB = bufA + (size_t)N * 128;          // [N,128] hidden   -> later h = attended
    float* bufC = bufB + (size_t)N * 128;          // [N,64]  Q (pre-scaled)
    float* bufD = bufC + (size_t)N * 64;           // [N,64]  ctx
    float* bufK = bufD + (size_t)N * 64;           // [128,64] K
    float* bufV = bufK + (size_t)L * 64;           // [128,64] V
    float* outp = (float*)d_out;                   // mu [N,64] ++ var [N,64]

    const dim3 blk(128);
    const int  gM = (N + 63) / 64;
    const int  gL = (L + 63) / 64;
    const float qscale = 0.17677669529663687f;     // (DK/HEADS)^-0.5 = 32^-0.5

    // 1) support1 = leaky(ns_emb @ W_hidden)
    gemm_wmma<128><<<gM, blk, 0, stream>>>(ns_emb, W_hidden, bufA, N, 256, 1, 1.0f);
    // 2) hidden = leaky(spmm(support1))
    hipMemsetAsync(bufB, 0, (size_t)N * 128 * sizeof(float), stream);
    spmm_kernel<<<2048, 256, 0, stream>>>(erows, ecols, evals, bufA, bufB, E, 128);
    leaky_kernel<<<2048, 256, 0, stream>>>(bufB, (size_t)N * 128);
    // 3) projections
    gemm_wmma<64><<<gM, blk, 0, stream>>>(bufB, Wq, bufC, N, 128, 0, qscale);
    gemm_wmma<64><<<gL, blk, 0, stream>>>(cond, Wk, bufK, L, 128, 0, 1.0f);
    gemm_wmma<64><<<gL, blk, 0, stream>>>(cond, Wv, bufV, L, 128, 0, 1.0f);
    // 4) attention -> ctx
    attn_kernel<<<(N + 31) / 32, 64, 0, stream>>>(bufC, bufK, bufV, bufD, N);
    // 5) h = ctx @ Wo   (hidden dead; reuse bufB)
    gemm_wmma<128><<<gM, blk, 0, stream>>>(bufD, Wo, bufB, N, 64, 0, 1.0f);
    // 6) mu / var supports (support1 dead; reuse bufA)
    gemm_wmma<64><<<gM, blk, 0, stream>>>(bufB, W_mu, bufA, N, 128, 1, 1.0f);
    gemm_wmma<64><<<gM, blk, 0, stream>>>(bufB, W_var, bufA + (size_t)N * 64, N, 128, 1, 1.0f);
    // 7) mu = leaky(spmm(support_mu)), var = leaky(spmm(support_var)) -> d_out
    hipMemsetAsync(outp, 0, (size_t)2 * N * 64 * sizeof(float), stream);
    spmm_kernel<<<2048, 256, 0, stream>>>(erows, ecols, evals, bufA, outp, E, 64);
    spmm_kernel<<<2048, 256, 0, stream>>>(erows, ecols, evals, bufA + (size_t)N * 64,
                                          outp + (size_t)N * 64, E, 64);
    leaky_kernel<<<2048, 256, 0, stream>>>(outp, (size_t)2 * N * 64);
}
